// SwiGLUMoETorch_15925738733694
// MI455X (gfx1250) — compile-verified, matrix operand order checked by reading, not run on hardware
//
#include <hip/hip_runtime.h>
#include <hip/hip_bf16.h>

// ---------------- problem constants ----------------
constexpr int BN    = 2;
constexpr int SN    = 2048;
constexpr int NTOK  = BN * SN;   // 4096 tokens
constexpr int HDIM  = 1024;
constexpr int IDIM  = 2048;
constexpr int ENUM  = 8;
constexpr int TOPK  = 2;
constexpr int NPAIR = NTOK * TOPK;           // 8192 routed (token,expert) pairs
constexpr int TILE_M = 128;                  // pairs per tile
constexpr int MAXTILES = NPAIR / TILE_M + ENUM; // 72 upper bound

typedef unsigned short u16;
typedef __attribute__((ext_vector_type(16))) __bf16 v16bf;
typedef __attribute__((ext_vector_type(8)))  float  v8f;

struct TileDesc { int e; int base; int rows; int pad; };

// ---------------- helpers ----------------
static __device__ __forceinline__ u16 f32_to_bf16(float f) {
    unsigned int u = __float_as_uint(f);
    unsigned int r = (u + 0x7FFFu + ((u >> 16) & 1u)) >> 16;   // RNE
    return (u16)r;
}

// Reinterpret two 16B register quads as one 16-element bf16 fragment.
static __device__ __forceinline__ v16bf make_frag(uint4 lo, uint4 hi) {
    union { uint4 q[2]; v16bf f; } u;
    u.q[0] = lo; u.q[1] = hi;
    return u.f;
}

// Fragment layout (ISA 7.12.2, 16-bit 16x32): lane = (half hf, index rc);
// element j=0..7  <-> k = hf*8 + j        (VGPRs 0-3)
// element j=8..15 <-> k = 16 + hf*8 + j-8 (VGPRs 4-7)
// => two contiguous 8xbf16 (16B) runs at k-offsets hf*8 and 16+hf*8.
static __device__ __forceinline__ v16bf load_frag_row(const u16* row, int hf) {
    uint4 lo = *(const uint4*)(row + hf * 8);
    uint4 hi = *(const uint4*)(row + 16 + hf * 8);
    return make_frag(lo, hi);
}

// ---------------- K1: fp32 -> bf16 conversion (vectorized) ----------------
__global__ __launch_bounds__(256) void cvt_bf16(const float4* __restrict__ in,
                                                u16* __restrict__ out, size_t n4) {
    size_t i = (size_t)blockIdx.x * blockDim.x + threadIdx.x;
    size_t stride = (size_t)gridDim.x * blockDim.x;
    for (; i < n4; i += stride) {
        float4 v = in[i];
        unsigned int lo = (unsigned int)f32_to_bf16(v.x) | ((unsigned int)f32_to_bf16(v.y) << 16);
        unsigned int hi = (unsigned int)f32_to_bf16(v.z) | ((unsigned int)f32_to_bf16(v.w) << 16);
        ((uint2*)out)[i] = make_uint2(lo, hi);
    }
}

// ---------------- K2: router (one wave per token) ----------------
__global__ __launch_bounds__(256) void router_kernel(
    const float* __restrict__ x, const float* __restrict__ Wg,
    int* __restrict__ e0, int* __restrict__ e1,
    float* __restrict__ w0, float* __restrict__ w1,
    int* __restrict__ counts)
{
    int token = blockIdx.x * 8 + (threadIdx.x >> 5);
    int lane  = threadIdx.x & 31;
    if (token >= NTOK) return;

    float acc[ENUM];
    #pragma unroll
    for (int e = 0; e < ENUM; ++e) acc[e] = 0.f;

    const float* xr = x + (size_t)token * HDIM;
    for (int h = lane; h < HDIM; h += 32) {
        float xv = xr[h];
        const float* wr = Wg + (size_t)h * ENUM;   // row-major [H][E]
        #pragma unroll
        for (int e = 0; e < ENUM; ++e) acc[e] += xv * wr[e];
    }
    #pragma unroll
    for (int e = 0; e < ENUM; ++e) {
        #pragma unroll
        for (int off = 16; off > 0; off >>= 1)
            acc[e] += __shfl_xor(acc[e], off, 32);
    }
    if (lane == 0) {
        float mx = acc[0];
        #pragma unroll
        for (int e = 1; e < ENUM; ++e) mx = fmaxf(mx, acc[e]);
        float ex[ENUM], sum = 0.f;
        #pragma unroll
        for (int e = 0; e < ENUM; ++e) { ex[e] = __expf(acc[e] - mx); sum += ex[e]; }
        int b0 = 0; float v0 = ex[0];
        #pragma unroll
        for (int e = 1; e < ENUM; ++e) if (ex[e] > v0) { v0 = ex[e]; b0 = e; }
        int b1 = -1; float v1 = -1.f;
        #pragma unroll
        for (int e = 0; e < ENUM; ++e) if (e != b0 && ex[e] > v1) { v1 = ex[e]; b1 = e; }
        float p0 = v0 / sum, p1 = v1 / sum;
        float rs = p0 + p1 + 1e-8f;
        e0[token] = b0;  w0[token] = p0 / rs;
        e1[token] = b1;  w1[token] = p1 / rs;
        atomicAdd(&counts[b0], 1);
        atomicAdd(&counts[b1], 1);
    }
}

// ---------------- K3: offsets + tile table ----------
__global__ void setup_kernel(const int* __restrict__ counts,
                             int* __restrict__ base, int* __restrict__ fill,
                             TileDesc* __restrict__ tiles)
{
    if (threadIdx.x != 0 || blockIdx.x != 0) return;
    int b = 0, t = 0;
    for (int e = 0; e < ENUM; ++e) {
        base[e] = b; fill[e] = 0;
        int c = counts[e];
        for (int r = 0; r < c; r += TILE_M) {
            tiles[t].e = e; tiles[t].base = b + r;
            int rem = c - r;
            tiles[t].rows = rem < TILE_M ? rem : TILE_M;
            tiles[t].pad = 0; ++t;
        }
        b += c;
    }
    for (; t < MAXTILES; ++t) { tiles[t].e = 0; tiles[t].base = 0; tiles[t].rows = 0; tiles[t].pad = 0; }
}

// ---------------- K4: scatter into per-expert pair lists -----------
__global__ __launch_bounds__(256) void scatter_kernel(
    const int* __restrict__ e0, const int* __restrict__ e1,
    const float* __restrict__ w0, const float* __restrict__ w1,
    const int* __restrict__ base, int* __restrict__ fill,
    int* __restrict__ pair_token, float* __restrict__ pair_w)
{
    int token = blockIdx.x * 256 + threadIdx.x;
    if (token >= NTOK) return;
    int a = e0[token];
    int s = atomicAdd(&fill[a], 1);
    pair_token[base[a] + s] = token;  pair_w[base[a] + s] = w0[token];
    int b = e1[token];
    s = atomicAdd(&fill[b], 1);
    pair_token[base[b] + s] = token;  pair_w[base[b] + s] = w1[token];
}

// Load one 32(K) x 64(N) bf16 weight tile, TRANSPOSED into LDS Bs[n][k].
// Per thread: column n = tid&63, k-block kb = tid>>6 (8 consecutive k).
// Global reads are lane-coalesced u16 (32 lanes -> 64B); LDS write is one b128.
static __device__ __forceinline__ void load_B_transposed(
    u16 (*Bs)[40], const u16* __restrict__ gbase, size_t rowStride,
    int tid)
{
    int n  = tid & 63;
    int kb = tid >> 6;               // 0..3
    const u16* gp = gbase + (size_t)(kb * 8) * rowStride + n;
    unsigned int q0, q1, q2, q3;
    q0 = (unsigned int)gp[0 * rowStride] | ((unsigned int)gp[1 * rowStride] << 16);
    q1 = (unsigned int)gp[2 * rowStride] | ((unsigned int)gp[3 * rowStride] << 16);
    q2 = (unsigned int)gp[4 * rowStride] | ((unsigned int)gp[5 * rowStride] << 16);
    q3 = (unsigned int)gp[6 * rowStride] | ((unsigned int)gp[7 * rowStride] << 16);
    *(uint4*)(&Bs[n][kb * 8]) = make_uint4(q0, q1, q2, q3);
}

// ---------------- K6: gathered GEMM1 + SwiGLU (bf16 WMMA) -----------------
// tile: 128 pairs x 64 I-cols, K over H in chunks of 32. 8 waves; each wave
// owns a 16-row slice with 4 gate + 4 value 16x16 f32 accumulators.
// All 8 B fragments are staged into registers BEFORE the WMMA burst so the
// scheduler can overlap v_wmma with outstanding ds_loads (partial dscnt waits).
__global__ __launch_bounds__(256) void moe_gemm1(
    const u16* __restrict__ xb,  const u16* __restrict__ W11b,
    const u16* __restrict__ W12b,
    const int* __restrict__ pair_token, const float* __restrict__ pair_w,
    const TileDesc* __restrict__ tiles, u16* __restrict__ fused)
{
    __shared__ u16  Xs[TILE_M][40];    // A tile, row-major [row][k]
    __shared__ u16  B1s[64][40];       // B tile TRANSPOSED: [n][k]
    __shared__ u16  B2s[64][40];
    __shared__ int   tokS[TILE_M];
    __shared__ float pwS[TILE_M];

    TileDesc td = tiles[blockIdx.y];
    if (td.rows == 0) return;
    const int n0   = blockIdx.x * 64;
    const int tid  = threadIdx.x;
    const int lane = tid & 31;
    const int wv   = tid >> 5;

    if (tid < TILE_M) {
        if (tid < td.rows) { tokS[tid] = pair_token[td.base + tid]; pwS[tid] = pair_w[td.base + tid]; }
        else               { tokS[tid] = -1;                        pwS[tid] = 0.f; }
    }
    __syncthreads();

    const v8f vzero = {0.f,0.f,0.f,0.f,0.f,0.f,0.f,0.f};
    v8f accG[4], accV[4];
    #pragma unroll
    for (int t = 0; t < 4; ++t) { accG[t] = vzero; accV[t] = vzero; }

    const size_t wbase = (size_t)td.e * HDIM * IDIM;
    const int hf = lane >> 4;       // half-wave select
    const int rc = lane & 15;       // row (A) / col (B)

    for (int kk = 0; kk < HDIM; kk += 32) {
        // A tile: gather 128 rows x 32 bf16 (b128 in, b128 out)
        for (int idx = tid; idx < 512; idx += 256) {
            int r = idx >> 2, s = idx & 3;
            uint4 val = make_uint4(0u, 0u, 0u, 0u);
            int t = tokS[r];
            if (t >= 0) val = *(const uint4*)(xb + (size_t)t * HDIM + kk + s * 8);
            *(uint4*)(&Xs[r][s * 8]) = val;
        }
        // B tiles transposed into LDS
        const u16* g1 = W11b + wbase + (size_t)kk * IDIM + n0;
        const u16* g2 = W12b + wbase + (size_t)kk * IDIM + n0;
        load_B_transposed(B1s, g1, IDIM, tid);
        load_B_transposed(B2s, g2, IDIM, tid);
        if (kk + 32 < HDIM) {   // prefetch next K-chunk (global_prefetch_b8)
            __builtin_prefetch(g1 + (size_t)32 * IDIM + (tid & 63), 0, 1);
            __builtin_prefetch(g2 + (size_t)32 * IDIM + (tid & 63), 0, 1);
        }
        __syncthreads();

        // stage ALL fragments first, then issue the WMMA burst
        v16bf a = load_frag_row(&Xs[wv * 16 + rc][0], hf);
        v16bf bf1[4], bf2[4];
        #pragma unroll
        for (int t = 0; t < 4; ++t) {
            bf1[t] = load_frag_row(&B1s[t * 16 + rc][0], hf);
            bf2[t] = load_frag_row(&B2s[t * 16 + rc][0], hf);
        }
        #pragma unroll
        for (int t = 0; t < 4; ++t) {
            accG[t] = __builtin_amdgcn_wmma_f32_16x16x32_bf16(false, a, false, bf1[t], (short)0, accG[t], false, false);
            accV[t] = __builtin_amdgcn_wmma_f32_16x16x32_bf16(false, a, false, bf2[t], (short)0, accV[t], false, false);
        }
        __syncthreads();
    }

    // fused = SiLU(gate) * value * route_weight, stored bf16
    const int col = lane & 15;
    const int rb  = (lane >> 4) * 8;
    #pragma unroll
    for (int t = 0; t < 4; ++t) {
        #pragma unroll
        for (int r = 0; r < 8; ++r) {
            int row = wv * 16 + rb + r;
            if (row < td.rows) {
                float g = accG[t][r], vv = accV[t][r];
                float s = g / (1.0f + __expf(-g));
                float f = s * vv * pwS[row];
                fused[(size_t)(td.base + row) * IDIM + n0 + t * 16 + col] = f32_to_bf16(f);
            }
        }
    }
}

// ---------------- K7: GEMM2 (fused @ W2) + atomic scatter to out ----------
__global__ __launch_bounds__(256) void moe_gemm2(
    const u16* __restrict__ fused, const u16* __restrict__ W2b,
    const int* __restrict__ pair_token, const TileDesc* __restrict__ tiles,
    float* __restrict__ out)
{
    __shared__ u16 As[TILE_M][40];
    __shared__ u16 Bs[64][40];         // transposed [h][k]
    __shared__ int tokS[TILE_M];

    TileDesc td = tiles[blockIdx.y];
    if (td.rows == 0) return;
    const int h0   = blockIdx.x * 64;
    const int tid  = threadIdx.x;
    const int lane = tid & 31;
    const int wv   = tid >> 5;

    if (tid < TILE_M)
        tokS[tid] = (tid < td.rows) ? pair_token[td.base + tid] : -1;
    __syncthreads();

    const v8f vzero = {0.f,0.f,0.f,0.f,0.f,0.f,0.f,0.f};
    v8f acc[4];
    #pragma unroll
    for (int t = 0; t < 4; ++t) acc[t] = vzero;

    const size_t wbase = (size_t)td.e * IDIM * HDIM;
    const int hf = lane >> 4;
    const int rc = lane & 15;

    for (int kk = 0; kk < IDIM; kk += 32) {
        for (int idx = tid; idx < 512; idx += 256) {
            int r = idx >> 2, s = idx & 3;
            uint4 val = make_uint4(0u, 0u, 0u, 0u);
            if (r < td.rows) val = *(const uint4*)(fused + (size_t)(td.base + r) * IDIM + kk + s * 8);
            *(uint4*)(&As[r][s * 8]) = val;
        }
        const u16* g = W2b + wbase + (size_t)kk * HDIM + h0;
        load_B_transposed(Bs, g, HDIM, tid);
        if (kk + 32 < IDIM)
            __builtin_prefetch(g + (size_t)32 * HDIM + (tid & 63), 0, 1);
        __syncthreads();

        v16bf a = load_frag_row(&As[wv * 16 + rc][0], hf);
        v16bf bf[4];
        #pragma unroll
        for (int t = 0; t < 4; ++t)
            bf[t] = load_frag_row(&Bs[t * 16 + rc][0], hf);
        #pragma unroll
        for (int t = 0; t < 4; ++t)
            acc[t] = __builtin_amdgcn_wmma_f32_16x16x32_bf16(false, a, false, bf[t], (short)0, acc[t], false, false);
        __syncthreads();
    }

    const int col = lane & 15;
    const int rb  = (lane >> 4) * 8;
    #pragma unroll
    for (int t = 0; t < 4; ++t) {
        #pragma unroll
        for (int r = 0; r < 8; ++r) {
            int row = wv * 16 + rb + r;
            if (row < td.rows) {
                int tok = tokS[row];
                atomicAdd(&out[(size_t)tok * HDIM + h0 + t * 16 + col], acc[t][r]);
            }
        }
    }
}

// ---------------- host side ----------------
static inline size_t align256(size_t x) { return (x + 255) & ~(size_t)255; }

extern "C" void kernel_launch(void* const* d_in, const int* in_sizes, int n_in,
                              void* d_out, int out_size, void* d_ws, size_t ws_size,
                              hipStream_t stream) {
    (void)in_sizes; (void)n_in; (void)out_size; (void)ws_size;

    const float* x   = (const float*)d_in[0];   // [B,S,H]
    const float* Wg  = (const float*)d_in[1];   // [H,E]
    const float* W11 = (const float*)d_in[2];   // [E,H,I]
    const float* W12 = (const float*)d_in[3];   // [E,H,I]
    const float* W2  = (const float*)d_in[4];   // [E,I,H]
    float* out = (float*)d_out;                 // [B,S,H]

    // workspace layout
    char* ws = (char*)d_ws;
    size_t off = 0;
    const size_t wElems = (size_t)ENUM * HDIM * IDIM;      // 16.78M per weight
    u16* W11b  = (u16*)(ws + off); off = align256(off + wElems * 2);
    u16* W12b  = (u16*)(ws + off); off = align256(off + wElems * 2);
    u16* W2b   = (u16*)(ws + off); off = align256(off + wElems * 2);
    u16* xb    = (u16*)(ws + off); off = align256(off + (size_t)NTOK * HDIM * 2);
    u16* fused = (u16*)(ws + off); off = align256(off + (size_t)NPAIR * IDIM * 2);
    int* pair_token = (int*)(ws + off); off = align256(off + NPAIR * 4);
    float* pair_w   = (float*)(ws + off); off = align256(off + NPAIR * 4);
    int* e0 = (int*)(ws + off);   off = align256(off + NTOK * 4);
    int* e1 = (int*)(ws + off);   off = align256(off + NTOK * 4);
    float* w0 = (float*)(ws + off); off = align256(off + NTOK * 4);
    float* w1 = (float*)(ws + off); off = align256(off + NTOK * 4);
    int* counts = (int*)(ws + off); off = align256(off + ENUM * 4);
    int* base   = (int*)(ws + off); off = align256(off + ENUM * 4);
    int* fill   = (int*)(ws + off); off = align256(off + ENUM * 4);
    TileDesc* tiles = (TileDesc*)(ws + off); off = align256(off + MAXTILES * sizeof(TileDesc));

    // 0) zero output + counters (graph-capturable memset nodes)
    hipMemsetAsync(out, 0, (size_t)NTOK * HDIM * sizeof(float), stream);
    hipMemsetAsync(counts, 0, ENUM * sizeof(int), stream);

    // 1) fp32 -> bf16 (bf16 weights = 100MB, fits the 192MB L2 for K-loop reuse)
    cvt_bf16<<<4096, 256, 0, stream>>>((const float4*)W11, W11b, wElems / 4);
    cvt_bf16<<<4096, 256, 0, stream>>>((const float4*)W12, W12b, wElems / 4);
    cvt_bf16<<<4096, 256, 0, stream>>>((const float4*)W2,  W2b,  wElems / 4);
    cvt_bf16<<<2048, 256, 0, stream>>>((const float4*)x,   xb,   (size_t)NTOK * HDIM / 4);

    // 2) router: softmax + top-2 + renorm + per-expert counts
    router_kernel<<<NTOK / 8, 256, 0, stream>>>(x, Wg, e0, e1, w0, w1, counts);

    // 3) offsets + tile table
    setup_kernel<<<1, 1, 0, stream>>>(counts, base, fill, tiles);

    // 4) scatter into per-expert compact pair lists
    scatter_kernel<<<NTOK / 256, 256, 0, stream>>>(e0, e1, w0, w1, base, fill, pair_token, pair_w);

    // 5) gathered GEMM1 + SwiGLU  (grid: 32 I-tiles x 72 pair-tiles)
    moe_gemm1<<<dim3(IDIM / 64, MAXTILES), 256, 0, stream>>>(
        xb, W11b, W12b, pair_token, pair_w, tiles, fused);

    // 6) GEMM2 + atomic combine into out  (grid: 16 H-tiles x 72 pair-tiles)
    moe_gemm2<<<dim3(HDIM / 64, MAXTILES), 256, 0, stream>>>(
        fused, W2b, pair_token, tiles, out);
}
